// TriModalBlock_13967233646891
// MI455X (gfx1250) — compile-verified, matrix operand order checked by reading, not run on hardware
//
#include <hip/hip_runtime.h>

// ---------------- problem constants ----------------
constexpr int NH  = 16;      // heads
constexpr int DM  = 1024;    // model dim
constexpr int FF  = 4096;    // ffn dim
constexpr int NB  = 4;       // batch
constexpr int SL  = 1024;    // seq len
constexpr int MR  = NB * SL; // 4096 rows
constexpr int HD  = 64;      // head dim

typedef __bf16 bf16_t;
typedef bf16_t v16bf __attribute__((ext_vector_type(16)));
typedef float  v8f   __attribute__((ext_vector_type(8)));

union FragU { v16bf v; unsigned u[8]; };

__device__ __forceinline__ unsigned short f2bf(float x) {
  union { float f; unsigned u; } c; c.f = x;
  unsigned r = c.u + 0x7FFFu + ((c.u >> 16) & 1u);   // round-to-nearest-even
  return (unsigned short)(r >> 16);
}

// 16x32 bf16 A-fragment (or symmetric 32x16 B-fragment) from K-contiguous
// [line][k] LDS. CDNA5 16-bit layout:
//   lanes 0-15  : line=lane,    K = 0..7 (v0..3), 16..23 (v4..7)
//   lanes 16-31 : line=lane-16, K = 8..15 (v0..3), 24..31 (v4..7)
// -> two 16B-contiguous runs per lane: compiler merges to 2x ds_load_b128.
__device__ __forceinline__ v16bf frag_ld(const unsigned short* p, int line,
                                         int stride, int kbase, int lane) {
  const int half = lane >> 4;
  const unsigned* q = (const unsigned*)(p + line * stride + kbase);
  FragU f;
#pragma unroll
  for (int i = 0; i < 4; ++i) {
    f.u[i]     = q[half * 4 + i];
    f.u[4 + i] = q[8 + half * 4 + i];
  }
  return f.v;
}

__device__ __forceinline__ v8f wmma_bf16(v16bf a, v16bf b, v8f c) {
  return __builtin_amdgcn_wmma_f32_16x16x32_bf16(false, a, false, b,
                                                 (short)0, c, false, false);
}

// ---------------- weight f32 -> bf16 ----------------
__global__ void k_f32_to_bf16(const float* __restrict__ src,
                              unsigned short* __restrict__ dst, int n) {
  for (int i = blockIdx.x * blockDim.x + threadIdx.x; i < n;
       i += gridDim.x * blockDim.x)
    dst[i] = f2bf(src[i]);
}

// ---------------- WMMA GEMM: C[M,N] = act(A[M,K] @ Bw[K,N] + bias) ---------
// Block tile 128x128, 8 waves x (32x64), K-step 32, double-buffered LDS.
// ABF : A operand is bf16 (1) or f32 converted on stage (0)
// MODE: 0 = f32 [M,N], 1 = bf16 [M,N], 2 = bf16 QKV scatter [B,H,L,64]
template <int ABF, int MODE, int RELU, int HB>
__global__ __launch_bounds__(256)
void k_gemm(const void* __restrict__ Ap,
            const unsigned short* __restrict__ Bw,
            const float* __restrict__ bias,
            int N, int K, void* __restrict__ Out) {
  __shared__ unsigned short As[2][128 * 32];  // [row][k]
  __shared__ unsigned short Bs[2][128 * 32];  // [n][k] (K-contiguous)

  const int tid  = threadIdx.x;
  const int lane = tid & 31;
  const int w    = tid >> 5;
  const int waveM = w >> 1;       // 0..3 -> 32-row chunk
  const int waveN = w & 1;        // 0..1 -> 64-col chunk
  const int ln   = lane & 15;
  const int half = lane >> 4;
  const int m0 = blockIdx.x * 128;
  const int n0 = blockIdx.y * 128;

  const float*          Af = (const float*)Ap;
  const unsigned short* Ab = (const unsigned short*)Ap;

  v8f acc[2][4];
#pragma unroll
  for (int i = 0; i < 2; ++i)
#pragma unroll
    for (int j = 0; j < 4; ++j)
#pragma unroll
      for (int v = 0; v < 8; ++v) acc[i][j][v] = 0.f;

  unsigned short aS[16], bS[16];

  auto stage = [&](int k0) {
#pragma unroll
    for (int e = 0; e < 4; ++e) {           // A tile: 128x32
      int flat = tid * 4 + e * 1024;
      int r = flat >> 5, c = flat & 31;
      if (ABF) {
        *(ushort4*)&aS[e * 4] =
            *(const ushort4*)&Ab[(size_t)(m0 + r) * K + (k0 + c)];
      } else {
        float4 f = *(const float4*)&Af[(size_t)(m0 + r) * K + (k0 + c)];
        aS[e * 4 + 0] = f2bf(f.x); aS[e * 4 + 1] = f2bf(f.y);
        aS[e * 4 + 2] = f2bf(f.z); aS[e * 4 + 3] = f2bf(f.w);
      }
    }
#pragma unroll
    for (int e = 0; e < 4; ++e) {           // B tile: 32x128
      int flat = tid * 4 + e * 1024;
      int k = flat >> 7, n = flat & 127;
      *(ushort4*)&bS[e * 4] =
          *(const ushort4*)&Bw[(size_t)(k0 + k) * N + (n0 + n)];
    }
    if (k0 + 64 < K)                        // gfx1250 global_prefetch path
      __builtin_prefetch(&Bw[(size_t)(k0 + 64) * N + n0 + (tid & 127)], 0, 1);
  };

  auto commit = [&](int cur) {
#pragma unroll
    for (int e = 0; e < 4; ++e) {
      int flat = tid * 4 + e * 1024;
      *(uint2*)&As[cur][flat] = *(uint2*)&aS[e * 4];
    }
#pragma unroll
    for (int e = 0; e < 4; ++e) {
      int flat = tid * 4 + e * 1024;
      int k = flat >> 7, n = flat & 127;
#pragma unroll
      for (int q = 0; q < 4; ++q) Bs[cur][(n + q) * 32 + k] = bS[e * 4 + q];
    }
  };

  stage(0);
  int cur = 0;
  for (int k0 = 0; k0 < K; k0 += 32) {
    commit(cur);
    if (k0 + 32 < K) stage(k0 + 32);        // next tile in flight
    __syncthreads();
    v16bf af[2], bf[4];
#pragma unroll
    for (int i = 0; i < 2; ++i)
      af[i] = frag_ld(&As[cur][0], waveM * 32 + i * 16 + ln, 32, 0, lane);
#pragma unroll
    for (int j = 0; j < 4; ++j)
      bf[j] = frag_ld(&Bs[cur][0], waveN * 64 + j * 16 + ln, 32, 0, lane);
#pragma unroll
    for (int i = 0; i < 2; ++i)
#pragma unroll
      for (int j = 0; j < 4; ++j)
        acc[i][j] = wmma_bf16(af[i], bf[j], acc[i][j]);
    cur ^= 1;
  }

  float bj[4];
#pragma unroll
  for (int j = 0; j < 4; ++j)
    bj[j] = HB ? bias[n0 + waveN * 64 + j * 16 + ln] : 0.f;

  // C layout: lane = col, VGPR v = row v (lanes 0-15) / v+8 (lanes 16-31)
#pragma unroll
  for (int i = 0; i < 2; ++i)
#pragma unroll
    for (int j = 0; j < 4; ++j)
#pragma unroll
      for (int v = 0; v < 8; ++v) {
        int gm = m0 + waveM * 32 + i * 16 + v + half * 8;
        int gn = n0 + waveN * 64 + j * 16 + ln;
        float val = acc[i][j][v] + bj[j];
        if (RELU) val = fmaxf(val, 0.f);
        if (MODE == 0) {
          ((float*)Out)[(size_t)gm * N + gn] = val;
        } else if (MODE == 1) {
          ((unsigned short*)Out)[(size_t)gm * N + gn] = f2bf(val);
        } else {
          int bb = gm >> 10, l = gm & 1023, hh = gn >> 6, dd = gn & 63;
          ((unsigned short*)Out)[((((size_t)bb * NH + hh) << 10) + l) * HD + dd]
              = f2bf(val);
        }
      }
}

// ---------------- flash attention (bf16 WMMA, f32 accum) -------------------
__global__ __launch_bounds__(128)
void k_attn_wmma(const unsigned short* __restrict__ Qb,
                 const unsigned short* __restrict__ Kb,
                 const unsigned short* __restrict__ Vb,
                 const float* __restrict__ amask,
                 const int* __restrict__ pmask,
                 float* __restrict__ Out) {
  __shared__ unsigned short Qs[64 * 64];       // [q][d]
  __shared__ unsigned short Ks[64 * 64];       // [key][d]
  __shared__ unsigned short Vt[64 * 64];       // [d][key] (transposed V)
  __shared__ unsigned short Ps[4][16 * 64];    // per-wave P tile [q][key]

  const int tid  = threadIdx.x;
  const int lane = tid & 31;
  const int w    = tid >> 5;
  const int ln   = lane & 15;
  const int half = lane >> 4;
  const int q0 = blockIdx.x * 64;
  const int h  = blockIdx.y;
  const int b  = blockIdx.z;
  const size_t headoff = ((size_t)b * NH + h) * (size_t)(SL * HD);

#pragma unroll
  for (int e = 0; e < 8; ++e) {
    int flat = tid * 4 + e * 512;
    int r = flat >> 6, c = flat & 63;
    *(ushort4*)&Qs[flat] =
        *(const ushort4*)&Qb[headoff + (size_t)(q0 + r) * HD + c];
  }
  __syncthreads();

  v16bf qf[2];
  qf[0] = frag_ld(Qs, w * 16 + ln, 64, 0, lane);
  qf[1] = frag_ld(Qs, w * 16 + ln, 64, 32, lane);

  float m_[8], l_[8];
  v8f o[4];
#pragma unroll
  for (int v = 0; v < 8; ++v) { m_[v] = -1e30f; l_[v] = 0.f; }
#pragma unroll
  for (int nt = 0; nt < 4; ++nt)
#pragma unroll
    for (int v = 0; v < 8; ++v) o[nt][v] = 0.f;

  for (int kb = 0; kb < SL; kb += 64) {
    __syncthreads();
#pragma unroll
    for (int e = 0; e < 8; ++e) {
      int flat = tid * 4 + e * 512;
      int r = flat >> 6, c = flat & 63;
      *(ushort4*)&Ks[flat] =
          *(const ushort4*)&Kb[headoff + (size_t)(kb + r) * HD + c];
      ushort4 vv = *(const ushort4*)&Vb[headoff + (size_t)(kb + r) * HD + c];
      Vt[(c + 0) * 64 + r] = vv.x;
      Vt[(c + 1) * 64 + r] = vv.y;
      Vt[(c + 2) * 64 + r] = vv.z;
      Vt[(c + 3) * 64 + r] = vv.w;
    }
    __syncthreads();

    // S = Q @ K^T  (B-frag: B(kk,n) = Ks[n*64+kk], K-contiguous)
    v8f s[4];
#pragma unroll
    for (int nt = 0; nt < 4; ++nt) {
#pragma unroll
      for (int v = 0; v < 8; ++v) s[nt][v] = 0.f;
      v16bf kf0 = frag_ld(Ks, nt * 16 + ln, 64, 0, lane);
      v16bf kf1 = frag_ld(Ks, nt * 16 + ln, 64, 32, lane);
      s[nt] = wmma_bf16(qf[0], kf0, s[nt]);
      s[nt] = wmma_bf16(qf[1], kf1, s[nt]);
    }
    // scale + attn_mask + pad_mask (reference adds both to logits)
#pragma unroll
    for (int nt = 0; nt < 4; ++nt)
#pragma unroll
      for (int v = 0; v < 8; ++v) {
        int grow = q0 + w * 16 + v + half * 8;
        int key  = kb + nt * 16 + ln;
        s[nt][v] = s[nt][v] * 0.125f + amask[(size_t)grow * SL + key]
                 + (float)pmask[b * SL + key];
      }
    // online softmax; row stats live in one 16-lane half -> xor 1/2/4/8
#pragma unroll
    for (int v = 0; v < 8; ++v) {
      float mx = fmaxf(fmaxf(s[0][v], s[1][v]), fmaxf(s[2][v], s[3][v]));
      mx = fmaxf(mx, __shfl_xor(mx, 1, 32));
      mx = fmaxf(mx, __shfl_xor(mx, 2, 32));
      mx = fmaxf(mx, __shfl_xor(mx, 4, 32));
      mx = fmaxf(mx, __shfl_xor(mx, 8, 32));
      float nm   = fmaxf(m_[v], mx);
      float corr = __expf(m_[v] - nm);
      float sum = 0.f;
#pragma unroll
      for (int nt = 0; nt < 4; ++nt) {
        float p = __expf(s[nt][v] - nm);
        s[nt][v] = p; sum += p;
      }
      sum += __shfl_xor(sum, 1, 32);
      sum += __shfl_xor(sum, 2, 32);
      sum += __shfl_xor(sum, 4, 32);
      sum += __shfl_xor(sum, 8, 32);
      l_[v] = l_[v] * corr + sum;
      m_[v] = nm;
#pragma unroll
      for (int nt = 0; nt < 4; ++nt) o[nt][v] *= corr;
    }
    // C-layout -> A-layout transpose of P through per-wave LDS
    unsigned short* Pw = &Ps[w][0];
#pragma unroll
    for (int nt = 0; nt < 4; ++nt)
#pragma unroll
      for (int v = 0; v < 8; ++v)
        Pw[(v + half * 8) * 64 + nt * 16 + ln] = f2bf(s[nt][v]);
    __syncthreads();

    // O += P @ V  (B-frag: B(kk,n) = Vt[n*64+kk], K-contiguous)
    v16bf pf0 = frag_ld(Pw, ln, 64, 0, lane);
    v16bf pf1 = frag_ld(Pw, ln, 64, 32, lane);
#pragma unroll
    for (int nt = 0; nt < 4; ++nt) {
      v16bf vf0 = frag_ld(Vt, nt * 16 + ln, 64, 0, lane);
      v16bf vf1 = frag_ld(Vt, nt * 16 + ln, 64, 32, lane);
      o[nt] = wmma_bf16(pf0, vf0, o[nt]);
      o[nt] = wmma_bf16(pf1, vf1, o[nt]);
    }
  }

#pragma unroll
  for (int nt = 0; nt < 4; ++nt)
#pragma unroll
    for (int v = 0; v < 8; ++v) {
      int grow = q0 + w * 16 + v + half * 8;
      int col  = nt * 16 + ln;
      Out[((size_t)b * SL + grow) * DM + h * HD + col] = o[nt][v] / l_[v];
    }
}

// ---------------- residual + LayerNorm -------------------------------------
__global__ __launch_bounds__(256)
void k_add_ln(const float* __restrict__ x, const float* __restrict__ y,
              const float* __restrict__ g, const float* __restrict__ bta,
              float* __restrict__ out) {
  __shared__ float r1[256], r2[256];
  const int row = blockIdx.x;
  const int tid = threadIdx.x;
  const float* xr = x + (size_t)row * DM;
  const float* yr = y + (size_t)row * DM;
  float4 xv = *(const float4*)&xr[tid * 4];
  float4 yv = *(const float4*)&yr[tid * 4];
  float e0 = xv.x + yv.x, e1 = xv.y + yv.y, e2 = xv.z + yv.z, e3 = xv.w + yv.w;
  r1[tid] = e0 + e1 + e2 + e3;
  r2[tid] = e0 * e0 + e1 * e1 + e2 * e2 + e3 * e3;
  __syncthreads();
  for (int s = 128; s > 0; s >>= 1) {
    if (tid < s) { r1[tid] += r1[tid + s]; r2[tid] += r2[tid + s]; }
    __syncthreads();
  }
  float mean = r1[0] * (1.f / DM);
  float var  = r2[0] * (1.f / DM) - mean * mean;
  float rstd = rsqrtf(var + 1e-5f);
  float4 gv = *(const float4*)&g[tid * 4];
  float4 bv = *(const float4*)&bta[tid * 4];
  float4 ov;
  ov.x = (e0 - mean) * rstd * gv.x + bv.x;
  ov.y = (e1 - mean) * rstd * gv.y + bv.y;
  ov.z = (e2 - mean) * rstd * gv.z + bv.z;
  ov.w = (e3 - mean) * rstd * gv.w + bv.w;
  *(float4*)&out[(size_t)row * DM + tid * 4] = ov;
}

// ---------------- host orchestration ---------------------------------------
extern "C" void kernel_launch(void* const* d_in, const int* in_sizes, int n_in,
                              void* d_out, int out_size, void* d_ws, size_t ws_size,
                              hipStream_t stream) {
  (void)in_sizes; (void)n_in; (void)out_size; (void)ws_size;
  const float* T     = (const float*)d_in[0];
  const float* A     = (const float*)d_in[1];
  const float* amask = (const float*)d_in[2];
  const int*   pmask = (const int*)d_in[3];
  const float* Wq  = (const float*)d_in[4];
  const float* bq  = (const float*)d_in[5];
  const float* Wk  = (const float*)d_in[6];
  const float* bk  = (const float*)d_in[7];
  const float* Wv  = (const float*)d_in[8];
  const float* bv  = (const float*)d_in[9];
  const float* Wo  = (const float*)d_in[10];
  const float* fW1 = (const float*)d_in[11];
  const float* fb1 = (const float*)d_in[12];
  const float* fW2 = (const float*)d_in[13];
  const float* fb2 = (const float*)d_in[14];
  const float* lng = (const float*)d_in[15];
  const float* lnb = (const float*)d_in[16];

  const size_t DD = (size_t)DM * DM;   // 1M
  const size_t DF = (size_t)DM * FF;   // 4M
  const size_t MD = (size_t)MR * DM;   // 4M

  char* base = (char*)d_ws;
  size_t off = 0;
  auto carve = [&](size_t bytes) -> void* {
    void* p = base + off;
    off += (bytes + 255) & ~(size_t)255;
    return p;
  };
  unsigned short* WqB = (unsigned short*)carve(4 * DD * 2);
  unsigned short* WkB = (unsigned short*)carve(4 * DD * 2);
  unsigned short* WvB = (unsigned short*)carve(4 * DD * 2);
  unsigned short* WoB = (unsigned short*)carve(4 * DD * 2);
  unsigned short* W1B = (unsigned short*)carve(2 * DF * 2);
  unsigned short* W2B = (unsigned short*)carve(2 * DF * 2);
  unsigned short* Qb  = (unsigned short*)carve(MD * 2);
  unsigned short* Kb  = (unsigned short*)carve(MD * 2);
  unsigned short* Vb  = (unsigned short*)carve(MD * 2);
  float* attnout = (float*)carve(MD * 4);
  float* proj    = (float*)carve(MD * 4);
  float* T1      = (float*)carve(MD * 4);
  float* A1      = (float*)carve(MD * 4);
  float* T2      = (float*)carve(MD * 4);
  float* A2      = (float*)carve(MD * 4);
  unsigned short* hidden = (unsigned short*)carve((size_t)MR * FF * 2);

  auto cvt = [&](const float* s, unsigned short* d, size_t n) {
    k_f32_to_bf16<<<2048, 256, 0, stream>>>(s, d, (int)n);
  };
  cvt(Wq,  WqB, 4 * DD);
  cvt(Wk,  WkB, 4 * DD);
  cvt(Wv,  WvB, 4 * DD);
  cvt(Wo,  WoB, 4 * DD);
  cvt(fW1, W1B, 2 * DF);
  cvt(fW2, W2B, 2 * DF);

  const dim3 blk(256);
  const dim3 gD(MR / 128, DM / 128);   // N = 1024
  const dim3 gF(MR / 128, FF / 128);   // N = 4096

  auto mha = [&](int i, const float* xq, const float* xk, const float* xv,
                 const float* resid, float* outbuf) {
    // QKV projections: f32 A, bias, bf16 head-scatter output
    k_gemm<0, 2, 0, 1><<<gD, blk, 0, stream>>>(xq, WqB + i * DD, bq + i * DM,
                                               DM, DM, Qb);
    k_gemm<0, 2, 0, 1><<<gD, blk, 0, stream>>>(xk, WkB + i * DD, bk + i * DM,
                                               DM, DM, Kb);
    k_gemm<0, 2, 0, 1><<<gD, blk, 0, stream>>>(xv, WvB + i * DD, bv + i * DM,
                                               DM, DM, Vb);
    k_attn_wmma<<<dim3(SL / 64, NH, NB), dim3(128), 0, stream>>>(
        Qb, Kb, Vb, amask, pmask, attnout);
    // output projection: no bias, f32 out
    k_gemm<0, 0, 0, 0><<<gD, blk, 0, stream>>>(attnout, WoB + i * DD, nullptr,
                                               DM, DM, proj);
    k_add_ln<<<MR, blk, 0, stream>>>(resid, proj, lng + i * DM, lnb + i * DM,
                                     outbuf);
  };

  mha(0, T,  T,  T,  T,  T1);   // self-attn time  -> norm1_time
  mha(1, A,  A,  A,  A,  A1);   // self-attn act   -> norm1_act
  mha(2, T1, A1, A1, T1, T2);   // cross T<-A      -> norm2_time
  mha(3, A1, T2, T2, A1, A2);   // cross A<-T      -> norm2_act

  float* Tout = (float*)d_out;
  float* Aout = (float*)d_out + MD;
  // FFN time: hidden = relu(T2@W1+b1) -> bf16 ; Tout = hidden@W2+b2 -> f32
  k_gemm<0, 1, 1, 1><<<gF, blk, 0, stream>>>(T2, W1B, fb1, FF, DM, hidden);
  k_gemm<1, 0, 0, 1><<<gD, blk, 0, stream>>>(hidden, W2B, fb2, DM, FF, Tout);
  // FFN act
  k_gemm<0, 1, 1, 1><<<gF, blk, 0, stream>>>(A2, W1B + DF, fb1 + FF, FF, DM,
                                             hidden);
  k_gemm<1, 0, 0, 1><<<gD, blk, 0, stream>>>(hidden, W2B + DF, fb2 + FF, DM,
                                             FF, Aout);
}
